// VectorQuantizerEMA_25864293056920
// MI455X (gfx1250) — matrix operand order repaired and checked by reading, not compiled, not run on hardware
//
#include <hip/hip_runtime.h>
#include <hip/hip_bf16.h>
#include <math.h>
#include <stdint.h>

typedef __attribute__((ext_vector_type(16))) _Float16 v16h;
typedef __attribute__((ext_vector_type(8)))  float    v8f;

#define NROWS   131072      // B*H*W
#define CDIM    64
#define KCODES  512
#define HW      4096        // H*W
#define ROWS_PER_WG   128   // 4 waves x 32 rows
#define WAVE_ROWS     32    // two 16-row WMMA A-tiles per wave
#define NTILES  32          // KCODES/16

// ---------------- workspace layout (d_ws) ----------------
// [0      .. 65535 ] bfrag : _Float16[32*2*32*16]  packed WMMA B fragments
// [65536  .. 67583 ] e2    : float[512]            0.5*||e_k||^2
// [67584  .. 69631 ] counts: unsigned[512]
// [69632  .. 69635 ] acc   : float                 sum of (quant-x)^2

// Pack codebook into per-lane B-fragment layout for v_wmma_f32_16x16x32_f16.
// B tile = 32(K) x 16(codes). Lanes 0-15 -> code col = lane, K 0..15;
// lanes 16-31 -> code col = lane-16, K 16..31. Half i in v16h -> K = 16*(l/16)+i.
__global__ void vq_prep(const float* __restrict__ emb,
                        _Float16* __restrict__ bfrag,
                        float* __restrict__ e2,
                        unsigned* __restrict__ counts,
                        float* __restrict__ acc) {
    int tid = blockIdx.x * blockDim.x + threadIdx.x;   // 32768 threads
    if (tid < 32 * 2 * 32 * 16) {
        int i    = tid & 15;
        int lane = (tid >> 4) & 31;
        int blk  = (tid >> 9) & 1;
        int tile = tid >> 10;
        int k = tile * 16 + (lane & 15);
        int c = blk * 32 + ((lane >> 4) << 4) + i;
        bfrag[tid] = (_Float16)emb[k * CDIM + c];
    }
    if (tid < KCODES) {
        float s = 0.f;
        const float* row = emb + tid * CDIM;
        #pragma unroll 8
        for (int c = 0; c < CDIM; ++c) s += row[c] * row[c];
        e2[tid] = 0.5f * s;
        counts[tid] = 0u;
    }
    if (tid == 0) *acc = 0.f;
}

__global__ __launch_bounds__(128)
void vq_main(const float* __restrict__ x,
             const float* __restrict__ emb,
             const _Float16* __restrict__ bfrag,
             const float* __restrict__ e2,
             unsigned* __restrict__ counts,
             float* __restrict__ acc,
             float* __restrict__ out) {
    __shared__ float xt[ROWS_PER_WG][CDIM + 4];   // padded: ~34 KB
    __shared__ int   idx_s[ROWS_PER_WG];
    __shared__ float lred[4];

    const int tid  = threadIdx.x;
    const int wave = tid >> 5;
    const int lane = tid & 31;
    const int ln   = lane & 15;
    const int hi   = lane >> 4;            // 0 or 1 (upper lane half)

    const int R   = blockIdx.x * ROWS_PER_WG;    // first flat row
    const int bb  = R >> 12;                     // batch index (R / HW)
    const int hw0 = R & (HW - 1);
    const float* xbase = x + (size_t)bb * CDIM * HW + hw0;   // x[b][c][hw0+r]

    // ---- phase 1: NCHW -> row-major LDS tile via async global->LDS copies ----
    // One lane per row; per channel the 128 threads read 128 contiguous floats.
    {
        int r = tid;                                   // 0..127
        const float* gp = xbase + r;
        unsigned lds0 = (unsigned)(uintptr_t)&xt[r][0];  // LDS byte offset (addr[31:0])
        #pragma unroll 8
        for (int c = 0; c < CDIM; ++c) {
            asm volatile("global_load_async_to_lds_b32 %0, %1, off"
                         :: "v"(lds0 + 4u * c), "v"(gp + (size_t)c * HW)
                         : "memory");
        }
        asm volatile("s_wait_asynccnt 0x0" ::: "memory");
    }
    __syncthreads();

    // ---- phase 2: build A fragments (16-bit A 16x32 layout), 2 tiles/wave ----
    const int rw = wave * WAVE_ROWS;
    const int m0 = rw + ln;                // row for A-tile 0
    const int m1 = rw + 16 + ln;           // row for A-tile 1
    v16h a0, a1, a2, a3;
    #pragma unroll
    for (int i = 0; i < 16; ++i) {
        int ko = hi * 16 + i;   // K offset within 32-block
        a0[i] = (_Float16)xt[m0][ko];
        a1[i] = (_Float16)xt[m0][32 + ko];
        a2[i] = (_Float16)xt[m1][ko];
        a3[i] = (_Float16)xt[m1][32 + ko];
    }

    float minv0[8], minv1[8];
    int   mini0[8], mini1[8];
    #pragma unroll
    for (int v = 0; v < 8; ++v) {
        minv0[v] = 3.4e38f; mini0[v] = 0;
        minv1[v] = 3.4e38f; mini1[v] = 0;
    }

    // ---- WMMA loop: score = 0.5*||e||^2 + x.e, argmin over 512 codes ----
    for (int t = 0; t < NTILES; ++t) {
        const _Float16* bp0 = bfrag + ((size_t)(t * 2 + 0) * 32 + lane) * 16;
        const _Float16* bp1 = bfrag + ((size_t)(t * 2 + 1) * 32 + lane) * 16;
        if (t + 1 < NTILES)
            __builtin_prefetch(bfrag + ((size_t)(t * 2 + 2) * 32 + lane) * 16, 0, 3);
        v16h b0 = *(const v16h*)bp0;
        v16h b1 = *(const v16h*)bp1;
        float ev = e2[t * 16 + ln];
        v8f c0, c1;
        #pragma unroll
        for (int v = 0; v < 8; ++v) { c0[v] = ev; c1[v] = ev; }
        c0 = __builtin_amdgcn_wmma_f32_16x16x32_f16(false, a0, false, b0,
                                                    (short)0, c0, false, false);
        c1 = __builtin_amdgcn_wmma_f32_16x16x32_f16(false, a2, false, b0,
                                                    (short)0, c1, false, false);
        c0 = __builtin_amdgcn_wmma_f32_16x16x32_f16(false, a1, false, b1,
                                                    (short)0, c0, false, false);
        c1 = __builtin_amdgcn_wmma_f32_16x16x32_f16(false, a3, false, b1,
                                                    (short)0, c1, false, false);
        int code = t * 16 + ln;
        #pragma unroll
        for (int v = 0; v < 8; ++v) {
            if (c0[v] < minv0[v]) { minv0[v] = c0[v]; mini0[v] = code; }
            if (c1[v] < minv1[v]) { minv1[v] = c1[v]; mini1[v] = code; }
        }
    }

    // ---- cross-lane argmin within the 16-lane column groups ----
    #pragma unroll
    for (int off = 8; off >= 1; off >>= 1) {
        #pragma unroll
        for (int v = 0; v < 8; ++v) {
            float om0 = __shfl_xor(minv0[v], off, 32);
            int   oi0 = __shfl_xor(mini0[v], off, 32);
            if (om0 < minv0[v] || (om0 == minv0[v] && oi0 < mini0[v])) {
                minv0[v] = om0; mini0[v] = oi0;
            }
            float om1 = __shfl_xor(minv1[v], off, 32);
            int   oi1 = __shfl_xor(mini1[v], off, 32);
            if (om1 < minv1[v] || (om1 == minv1[v] && oi1 < mini1[v])) {
                minv1[v] = om1; mini1[v] = oi1;
            }
        }
    }
    if (ln == 0) {
        #pragma unroll
        for (int v = 0; v < 8; ++v) {
            idx_s[rw +      hi * 8 + v] = mini0[v];  // lanes 0/16: rows 0-7 / 8-15
            idx_s[rw + 16 + hi * 8 + v] = mini1[v];  // of each A-tile
        }
    }
    __syncthreads();

    // ---- phase 3: outputs ----
    float* oq     = out + 1;
    float* oenc   = out + 1 + (size_t)NROWS * CDIM;
    float* operp  = oenc + NROWS;
    float* ocodes = operp + 1;
    (void)operp;

    {
        int id = idx_s[tid];
        float fv = (float)id;
        oenc[R + tid]   = fv;
        ocodes[R + tid] = fv;
        atomicAdd(&counts[id], 1u);
    }

    float part = 0.f;
    {
        int r  = tid;
        int id = idx_s[r];
        const float* erow = emb + (size_t)id * CDIM;
        float* qb = oq + (size_t)bb * CDIM * HW + hw0 + r;
        #pragma unroll 4
        for (int c = 0; c < CDIM; ++c) {
            float e = erow[c];
            float d = e - xt[r][c];
            part += d * d;
            qb[(size_t)c * HW] = e;     // coalesced across r for each channel c
        }
    }
    #pragma unroll
    for (int off = 16; off >= 1; off >>= 1) part += __shfl_xor(part, off, 32);
    if (lane == 0) lred[wave] = part;
    __syncthreads();
    if (tid == 0) atomicAdd(acc, lred[0] + lred[1] + lred[2] + lred[3]);
}

__global__ void vq_final(const unsigned* __restrict__ counts,
                         const float* __restrict__ acc,
                         float* __restrict__ out) {
    __shared__ float s[KCODES];
    int k = threadIdx.x;
    float p = (float)counts[k] * (1.0f / (float)NROWS);
    s[k] = -p * logf(p + 1e-10f);
    __syncthreads();
    for (int off = 256; off >= 1; off >>= 1) {
        if (k < off) s[k] += s[k + off];
        __syncthreads();
    }
    if (k == 0) {
        out[0] = 0.25f * (*acc) / (float)((size_t)NROWS * CDIM);   // BETA * mean
        out[1 + (size_t)NROWS * CDIM + NROWS] = expf(s[0]);        // perplexity
    }
}

extern "C" void kernel_launch(void* const* d_in, const int* in_sizes, int n_in,
                              void* d_out, int out_size, void* d_ws, size_t ws_size,
                              hipStream_t stream) {
    (void)in_sizes; (void)n_in; (void)out_size; (void)ws_size;
    const float* x   = (const float*)d_in[0];   // (32,64,64,64) f32
    const float* emb = (const float*)d_in[1];   // (512,64) f32

    char* ws = (char*)d_ws;
    _Float16* bfrag  = (_Float16*)ws;                       // 64 KB
    float*    e2     = (float*)(ws + 65536);                // 2 KB
    unsigned* counts = (unsigned*)(ws + 65536 + 2048);      // 2 KB
    float*    acc    = (float*)(ws + 65536 + 2048 + 2048);  // 4 B
    float*    out    = (float*)d_out;

    vq_prep<<<128, 256, 0, stream>>>(emb, bfrag, e2, counts, acc);
    vq_main<<<NROWS / ROWS_PER_WG, 128, 0, stream>>>(x, emb, bfrag, e2, counts, acc, out);
    vq_final<<<1, KCODES, 0, stream>>>(counts, acc, out);
}
